// AdaptiveNoisingModule_61692910240214
// MI455X (gfx1250) — matrix operand order repaired and checked by reading, not compiled, not run on hardware
//
#include <hip/hip_runtime.h>
#include <hip/hip_bf16.h>
#include <stdint.h>

// Problem dims
#define B_ 8
#define C_ 1024
#define H_ 28
#define W_ 28
#define M_ (B_*H_*W_)   // 6272 query rows
#define N_ 16384        // memory bank rows
#define K_ 1024         // feature dim
#define EPSF 1e-8f

typedef __attribute__((ext_vector_type(16))) __bf16 v16bf;
typedef __attribute__((ext_vector_type(8)))  float  v8f;

union AccU { v8f v; float f[8]; };
union FragU { v16bf v; uint4 q[2]; };

// ---------------- helpers ----------------

__device__ __forceinline__ unsigned short f32_to_bf16_rne(float x) {
  unsigned u = __float_as_uint(x);
  u += 0x7fffu + ((u >> 16) & 1u);
  return (unsigned short)(u >> 16);
}

__device__ __forceinline__ unsigned long long shfl_xor_u64(unsigned long long v, int m) {
  unsigned lo = (unsigned)v, hi = (unsigned)(v >> 32);
  lo = __shfl_xor(lo, m, 32);
  hi = __shfl_xor(hi, m, 32);
  return ((unsigned long long)hi << 32) | lo;
}

__device__ __forceinline__ float block_reduce_sum_256(float v, float* sbuf) {
  const int t = threadIdx.x;
  for (int m = 16; m >= 1; m >>= 1) v += __shfl_xor(v, m, 32);
  if ((t & 31) == 0) sbuf[t >> 5] = v;
  __syncthreads();
  if (t == 0) {
    float s = 0.f;
    for (int i = 0; i < 8; ++i) s += sbuf[i];
    sbuf[0] = s;
  }
  __syncthreads();
  return sbuf[0];
}

__device__ __forceinline__ unsigned rotl32(unsigned x, int r) {
  return (x << r) | (x >> (32 - r));
}

// CDNA5 async global->LDS copy (ASYNCcnt-tracked), 16B per lane.
__device__ __forceinline__ void async_copy_b128(unsigned lds_addr, const void* gaddr) {
  asm volatile("global_load_async_to_lds_b128 %0, %1, off"
               :: "v"(lds_addr), "v"(gaddr)
               : "memory");
}
__device__ __forceinline__ void wait_async0() {
  asm volatile("s_wait_asynccnt 0x0" ::: "memory");
}

// threefry2x32, key = (0,1) (jax.random.key(1)); JAX iota split-half counters.
__device__ float threefry_normal(unsigned g) {
  const unsigned HALF = (unsigned)(((size_t)M_ * C_) / 2); // 3211264
  unsigned c0, c1; int useHi;
  if (g < HALF) { c0 = g; c1 = g + HALF; useHi = 0; }
  else          { c0 = g - HALF; c1 = g; useHi = 1; }
  const unsigned k0 = 0u, k1 = 1u, k2 = 0x1BD11BDBu; // k0^k1^0x1BD11BDA
  unsigned x0 = c0 + k0, x1 = c1 + k1;
#define TFR(r) { x0 += x1; x1 = rotl32(x1, (r)); x1 ^= x0; }
  TFR(13) TFR(15) TFR(26) TFR(6)  x0 += k1; x1 += k2 + 1u;
  TFR(17) TFR(29) TFR(16) TFR(24) x0 += k2; x1 += k0 + 2u;
  TFR(13) TFR(15) TFR(26) TFR(6)  x0 += k0; x1 += k1 + 3u;
  TFR(17) TFR(29) TFR(16) TFR(24) x0 += k1; x1 += k2 + 4u;
  TFR(13) TFR(15) TFR(26) TFR(6)  x0 += k2; x1 += k0 + 5u;
#undef TFR
  unsigned bits = useHi ? x1 : x0;
  float f = __uint_as_float((bits >> 9) | 0x3f800000u) - 1.0f; // [0,1)
  float u = 2.f * f - 1.f;                                     // (-1,1)
  // erfinv (Giles single-precision approximation)
  float w = -__logf((1.f - u) * (1.f + u));
  float p;
  if (w < 5.f) {
    w -= 2.5f;
    p = 2.81022636e-08f;
    p = fmaf(p, w, 3.43273939e-07f);
    p = fmaf(p, w, -3.5233877e-06f);
    p = fmaf(p, w, -4.39150654e-06f);
    p = fmaf(p, w, 0.00021858087f);
    p = fmaf(p, w, -0.00125372503f);
    p = fmaf(p, w, -0.00417768164f);
    p = fmaf(p, w, 0.246640727f);
    p = fmaf(p, w, 1.50140941f);
  } else {
    w = sqrtf(w) - 3.f;
    p = -0.000200214257f;
    p = fmaf(p, w, 0.000100950558f);
    p = fmaf(p, w, 0.00134934322f);
    p = fmaf(p, w, -0.00367342844f);
    p = fmaf(p, w, 0.00573950773f);
    p = fmaf(p, w, -0.0076224613f);
    p = fmaf(p, w, 0.00943887047f);
    p = fmaf(p, w, 1.00167406f);
    p = fmaf(p, w, 2.83297682f);
  }
  return 1.41421356237f * p * u;
}

// ---------------- kernels ----------------

// 1) memory_bank -> bf16 + row squared norms (one block per row)
__global__ __launch_bounds__(256)
void pack_mb_kernel(const float* __restrict__ mb,
                    unsigned short* __restrict__ mbb,
                    float* __restrict__ mbnorm) {
  __shared__ float sbuf[8];
  const int n = blockIdx.x;
  const float* row = mb + (size_t)n * K_;
  float s = 0.f;
  for (int c = threadIdx.x; c < K_; c += 256) {
    float v = row[c];
    s += v * v;
    mbb[(size_t)n * K_ + c] = f32_to_bf16_rne(v);
  }
  s = block_reduce_sum_256(s, sbuf);
  if (threadIdx.x == 0) mbnorm[n] = s;
}

// 2) features NCHW -> f (M,C) fp32 + bf16 + row squared norms
__global__ __launch_bounds__(256)
void pack_f_kernel(const float* __restrict__ feat,
                   float* __restrict__ f32,
                   unsigned short* __restrict__ fb,
                   float* __restrict__ fnorm) {
  __shared__ float sbuf[8];
  const int m = blockIdx.x;
  const int w = m % W_;
  const int h = (m / W_) % H_;
  const int b = m / (H_ * W_);
  float s = 0.f;
  for (int c = threadIdx.x; c < K_; c += 256) {
    float v = feat[(((size_t)b * C_ + c) * H_ + h) * W_ + w];
    f32[(size_t)m * K_ + c] = v;
    fb[(size_t)m * K_ + c] = f32_to_bf16_rne(v);
    s += v * v;
  }
  s = block_reduce_sum_256(s, sbuf);
  if (threadIdx.x == 0) fnorm[m] = s;
}

// 3) init per-row (dist,idx) packs
__global__ __launch_bounds__(256)
void init_minpack_kernel(unsigned long long* __restrict__ minpack) {
  int m = blockIdx.x * 256 + threadIdx.x;
  if (m < M_) minpack[m] = 0xFFFFFFFFFFFFFFFFull;
}

// 4) bf16 WMMA GEMM (f . mb^T) with fused dist + argmin epilogue.
//    Block tile 128(M) x 128(N), 8 waves: waveM 0..3 (32 rows), waveN 0..1 (64 cols),
//    8 WMMAs per wave per 32-wide K slab. Double-buffered LDS fed by async
//    global->LDS copies; fully branch-free steady-state (2 slabs per trip,
//    compile-time LDS buffer bases).
#define LDS_STRIDE 40                  // ushorts; 80B stride -> conflict-free b128
#define TILE_HALF (128 * LDS_STRIDE)   // ushorts per A (or B) buffer
// ushort-index bases:  A0 | B0 | A1 | B1
#define A0_ 0
#define B0_ TILE_HALF
#define A1_ (2 * TILE_HALF)
#define B1_ (3 * TILE_HALF)

template <int ABASE, int BBASE>
__device__ __forceinline__ void stage_slab(unsigned lds_base, int ch0,
                                           const unsigned short* __restrict__ fb,
                                           const unsigned short* __restrict__ mbb,
                                           int rowBase, int colBase, int kk) {
#pragma unroll
  for (int j = 0; j < 2; ++j) {
    const int chunk = ch0 + j;
    const int row = chunk >> 2;           // 0..127
    const int koff = (chunk & 3) * 8;     // 0,8,16,24 ushorts
    const unsigned loff = (unsigned)((row * LDS_STRIDE + koff) * 2);
    async_copy_b128(lds_base + (unsigned)(ABASE * 2) + loff,
                    fb + (size_t)(rowBase + row) * K_ + kk + koff);
    async_copy_b128(lds_base + (unsigned)(BBASE * 2) + loff,
                    mbb + (size_t)(colBase + row) * K_ + kk + koff);
  }
}

template <int ABASE, int BBASE>
__device__ __forceinline__ void compute_slab(const unsigned short* lds,
                                             int waveM, int waveN, int r16, int hi,
                                             AccU (&acc)[2][4]) {
  FragU a[2];
#pragma unroll
  for (int ms = 0; ms < 2; ++ms) {
    const int arow = waveM * 32 + ms * 16 + r16;
    a[ms].q[0] = *(const uint4*)&lds[ABASE + arow * LDS_STRIDE + hi * 8];
    a[ms].q[1] = *(const uint4*)&lds[ABASE + arow * LDS_STRIDE + 16 + hi * 8];
  }
#pragma unroll
  for (int ns = 0; ns < 4; ++ns) {
    const int bcol = waveN * 64 + ns * 16 + r16;
    FragU b;
    b.q[0] = *(const uint4*)&lds[BBASE + bcol * LDS_STRIDE + hi * 8];
    b.q[1] = *(const uint4*)&lds[BBASE + bcol * LDS_STRIDE + 16 + hi * 8];
    acc[0][ns].v = __builtin_amdgcn_wmma_f32_16x16x32_bf16(
        false, a[0].v, false, b.v, (short)0, acc[0][ns].v, false, false);
    acc[1][ns].v = __builtin_amdgcn_wmma_f32_16x16x32_bf16(
        false, a[1].v, false, b.v, (short)0, acc[1][ns].v, false, false);
  }
}

__global__ __launch_bounds__(256)
void knn_gemm_kernel(const unsigned short* __restrict__ fb,
                     const unsigned short* __restrict__ mbb,
                     const float* __restrict__ fnorm,
                     const float* __restrict__ mbnorm,
                     unsigned long long* __restrict__ minpack) {
  __shared__ __align__(16) unsigned short lds[4 * TILE_HALF];

  const int t     = threadIdx.x;
  const int lane  = t & 31;
  const int wave  = t >> 5;
  const int waveM = wave & 3;    // 32-row tile
  const int waveN = wave >> 2;   // 64-col tile
  const int rowBase = blockIdx.y * 128;
  const int colBase = blockIdx.x * 128;

  const int r16 = lane & 15;
  const int hi  = lane >> 4;
  const int ch0 = t * 2;

  const unsigned lds_base = (unsigned)(size_t)&lds[0];

  AccU acc[2][4];
#pragma unroll
  for (int ms = 0; ms < 2; ++ms)
#pragma unroll
    for (int ns = 0; ns < 4; ++ns)
      acc[ms][ns].v = (v8f){0.f,0.f,0.f,0.f,0.f,0.f,0.f,0.f};

  // prologue: slab 0 -> buf0, slab 32 -> buf1
  stage_slab<A0_, B0_>(lds_base, ch0, fb, mbb, rowBase, colBase, 0);
  stage_slab<A1_, B1_>(lds_base, ch0, fb, mbb, rowBase, colBase, 32);
  wait_async0();
  __syncthreads();

  // steady state: two 32-K slabs per trip, all stages unconditional
  for (int kk = 0; kk + 64 < K_; kk += 64) {   // kk = 0,64,...,896
    compute_slab<A0_, B0_>(lds, waveM, waveN, r16, hi, acc);
    wait_async0();          // buf1's staging complete
    __syncthreads();        // everyone done reading buf0
    stage_slab<A0_, B0_>(lds_base, ch0, fb, mbb, rowBase, colBase, kk + 64);
    compute_slab<A1_, B1_>(lds, waveM, waveN, r16, hi, acc);
    wait_async0();          // buf0's staging complete
    __syncthreads();        // everyone done reading buf1
    stage_slab<A1_, B1_>(lds_base, ch0, fb, mbb, rowBase, colBase, kk + 96);
  }
  // tail: buf0 = slab 960, buf1 = slab 992 (both staged)
  compute_slab<A0_, B0_>(lds, waveM, waveN, r16, hi, acc);
  wait_async0();
  __syncthreads();
  compute_slab<A1_, B1_>(lds, waveM, waveN, r16, hi, acc);

  // Epilogue: d2 = |f|^2 + |m|^2 - 2*dot ; dist = sqrt(max(d2,0)+eps)
  // pack (dist_bits<<32)|col -> min over 4 N-subtiles -> 16-lane min -> atomicMin
  float mbn[4];
  int cols[4];
#pragma unroll
  for (int ns = 0; ns < 4; ++ns) {
    cols[ns] = colBase + waveN * 64 + ns * 16 + r16;
    mbn[ns] = mbnorm[cols[ns]];
  }
#pragma unroll
  for (int ms = 0; ms < 2; ++ms) {
#pragma unroll
    for (int i = 0; i < 8; ++i) {
      const int r = rowBase + waveM * 32 + ms * 16 + i + hi * 8;
      const float fn = fnorm[r];
      unsigned long long p = 0xFFFFFFFFFFFFFFFFull;
#pragma unroll
      for (int ns = 0; ns < 4; ++ns) {
        const float d = fn + mbn[ns] - 2.f * acc[ms][ns].f[i];
        const float dist = sqrtf(fmaxf(d, 0.f) + EPSF);
        const unsigned long long pk =
            ((unsigned long long)__float_as_uint(dist) << 32) | (unsigned)cols[ns];
        p = pk < p ? pk : p;
      }
      for (int msk = 1; msk < 16; msk <<= 1) {
        unsigned long long q = shfl_xor_u64(p, msk);
        p = q < p ? q : p;
      }
      if (r16 == 0) atomicMin(&minpack[r], p);
    }
  }
}

// 5) influence[m] = mean|f - nearest| / (dmin + eps)
__global__ __launch_bounds__(256)
void influence_kernel(const float* __restrict__ f32,
                      const float* __restrict__ mb,
                      const unsigned long long* __restrict__ minpack,
                      float* __restrict__ influence) {
  __shared__ float sbuf[8];
  const int m = blockIdx.x;
  const unsigned long long p = minpack[m];
  const float dist = __uint_as_float((unsigned)(p >> 32));
  const unsigned idx = (unsigned)(p & 0xFFFFFFFFu);
  const float* frow = f32 + (size_t)m * K_;
  const float* nrow = mb + (size_t)idx * K_;
  float s = 0.f;
  for (int c = threadIdx.x; c < K_; c += 256)
    s += fabsf(frow[c] - nrow[c]);
  s = block_reduce_sum_256(s, sbuf);
  if (threadIdx.x == 0)
    influence[m] = (s / (float)K_) / (dist + EPSF);
}

// 6) global min/max of influence (single block)
__global__ __launch_bounds__(256)
void minmax_kernel(const float* __restrict__ influence, float* __restrict__ stats) {
  __shared__ float smin[8];
  __shared__ float smax[8];
  const int t = threadIdx.x;
  float lmin = 3.4e38f, lmax = -3.4e38f;
  for (int i = t; i < M_; i += 256) {
    float v = influence[i];
    lmin = fminf(lmin, v);
    lmax = fmaxf(lmax, v);
  }
  for (int m = 16; m >= 1; m >>= 1) {
    lmin = fminf(lmin, __shfl_xor(lmin, m, 32));
    lmax = fmaxf(lmax, __shfl_xor(lmax, m, 32));
  }
  if ((t & 31) == 0) { smin[t >> 5] = lmin; smax[t >> 5] = lmax; }
  __syncthreads();
  if (t == 0) {
    float mn = smin[0], mx = smax[0];
    for (int i = 1; i < 8; ++i) { mn = fminf(mn, smin[i]); mx = fmaxf(mx, smax[i]); }
    stats[0] = mn; stats[1] = mx;
  }
}

// 7) noised output in NCHW; grid (C/8, B*H); threads: w = t&31 (28 active), cl = t>>5
__global__ __launch_bounds__(256)
void noised_kernel(const float* __restrict__ f32,
                   const float* __restrict__ influence,
                   const float* __restrict__ stats,
                   float* __restrict__ out) {
  const int t = threadIdx.x;
  const int w = t & 31;
  if (w >= W_) return;
  const int cl = t >> 5;
  const int c = blockIdx.x * 8 + cl;
  const int bh = blockIdx.y;
  const int b = bh / H_;
  const int h = bh % H_;
  const int m = (b * H_ + h) * W_ + w;
  const float imin = stats[0], imax = stats[1];
  const float inf = influence[m];
  const float norm = (imax - imin > EPSF) ? (inf - imin) / (imax - imin) : 0.f;
  const float nstd = 0.01f + norm * 0.49f;
  const float fv = f32[(size_t)m * K_ + c];
  const float z = threefry_normal((unsigned)(m * C_ + c));
  out[(((size_t)b * C_ + c) * H_ + h) * W_ + w] = fv + z * nstd;
}

// 8) per-row influence_norm and noise_std outputs
__global__ __launch_bounds__(256)
void perrow_out_kernel(const float* __restrict__ influence,
                       const float* __restrict__ stats,
                       float* __restrict__ out_inf,
                       float* __restrict__ out_std) {
  int m = blockIdx.x * 256 + threadIdx.x;
  if (m < M_) {
    const float imin = stats[0], imax = stats[1];
    const float inf = influence[m];
    const float norm = (imax - imin > EPSF) ? (inf - imin) / (imax - imin) : 0.f;
    out_inf[m] = norm;
    out_std[m] = 0.01f + norm * 0.49f;
  }
}

// ---------------- launch ----------------

extern "C" void kernel_launch(void* const* d_in, const int* in_sizes, int n_in,
                              void* d_out, int out_size, void* d_ws, size_t ws_size,
                              hipStream_t stream) {
  const float* feat = (const float*)d_in[0];  // (8,1024,28,28)
  const float* mb   = (const float*)d_in[1];  // (16384,1024)
  float* out = (float*)d_out;

  // workspace layout (256B aligned)
  char* ws = (char*)d_ws;
  size_t off = 0;
  unsigned short* mbb = (unsigned short*)(ws + off); off += (size_t)N_ * K_ * 2;          // 33.6MB
  unsigned short* fb  = (unsigned short*)(ws + off); off += (size_t)M_ * K_ * 2;          // 12.8MB
  float* f32          = (float*)(ws + off);          off += (size_t)M_ * K_ * 4;          // 25.7MB
  float* mbnorm       = (float*)(ws + off);          off += (size_t)N_ * 4;
  float* fnorm        = (float*)(ws + off);          off += (size_t)M_ * 4;
  unsigned long long* minpack = (unsigned long long*)(ws + off); off += (size_t)M_ * 8;
  float* influence    = (float*)(ws + off);          off += (size_t)M_ * 4;
  float* stats        = (float*)(ws + off);          off += 256;
  (void)ws_size; (void)in_sizes; (void)n_in; (void)out_size;

  pack_mb_kernel<<<N_, 256, 0, stream>>>(mb, mbb, mbnorm);
  pack_f_kernel<<<M_, 256, 0, stream>>>(feat, f32, fb, fnorm);
  init_minpack_kernel<<<(M_ + 255) / 256, 256, 0, stream>>>(minpack);

  dim3 ggrid(N_ / 128, M_ / 128);  // (128, 49)
  knn_gemm_kernel<<<ggrid, 256, 0, stream>>>(fb, mbb, fnorm, mbnorm, minpack);

  influence_kernel<<<M_, 256, 0, stream>>>(f32, mb, minpack, influence);
  minmax_kernel<<<1, 256, 0, stream>>>(influence, stats);

  dim3 ngrid(C_ / 8, B_ * H_);   // (128, 224)
  noised_kernel<<<ngrid, 256, 0, stream>>>(f32, influence, stats, out);

  float* out_inf = out + (size_t)B_ * C_ * H_ * W_;
  float* out_std = out_inf + M_;
  perrow_out_kernel<<<(M_ + 255) / 256, 256, 0, stream>>>(influence, stats, out_inf, out_std);
}
